// EdgeConditionedGNN_42494406426959
// MI455X (gfx1250) — matrix operand order repaired and checked by reading, not compiled, not run on hardware
//
#include <hip/hip_runtime.h>
#include <hip/hip_bf16.h>

#define HD 64
#define EPSV 1e-5f

typedef __attribute__((ext_vector_type(16))) _Float16 v16h;
typedef __attribute__((ext_vector_type(8)))  _Float16 v8h;
typedef __attribute__((ext_vector_type(8)))  float    v8f;
typedef __attribute__((ext_vector_type(4)))  unsigned int u32x4;
typedef __attribute__((ext_vector_type(4)))  int i32x4;
typedef __attribute__((ext_vector_type(8)))  int i32x8;

union V16 { v16h v; v8h h[2]; _Float16 e[16]; };

#define KCH  66                     // total K chunks (64 outer-product + 2 bias)
#define GRP  6                      // K-chunks per staged LDS group
#define NGRP 11                     // 66 / 6
#define GROUP_HALVES (GRP * 32 * 64)        // 12288 f16 = 24576 B
#define GROUP_BYTES  (GROUP_HALVES * 2)
#define GROUP_UNITS  (GROUP_BYTES / 8)      // 3072 units of 8 bytes

// ---- degree ----
__global__ void k_deg(const int* __restrict__ dst, float* __restrict__ deg, int E) {
  int i = blockIdx.x * blockDim.x + threadIdx.x;
  if (i < E) atomicAdd(&deg[dst[i]], 1.0f);
}
__global__ void k_invdeg(float* __restrict__ deg, int N) {
  int i = blockIdx.x * blockDim.x + threadIdx.x;
  if (i < N) deg[i] = 1.0f / fmaxf(deg[i], 1.0f);
}

// ---- input projection: h = x @ proj_w + proj_b ; also f16 copy ----
__global__ void k_proj(const float* __restrict__ x, const float* __restrict__ w,
                       const float* __restrict__ b, float* __restrict__ h,
                       _Float16* __restrict__ hh, int N) {
  int i = blockIdx.x * blockDim.x + threadIdx.x;
  if (i >= N * HD) return;
  int n = i >> 6, f = i & 63;
  float acc = b[f];
#pragma unroll
  for (int c = 0; c < 8; ++c) acc += x[n * 8 + c] * w[c * 64 + f];
  h[i] = acc;
  hh[i] = (_Float16)acc;
}

// ---- build permuted f16 B matrix [2112,64] ----
__global__ void k_prep(const float* __restrict__ w2, const float* __restrict__ b2,
                       _Float16* __restrict__ Bm, int l) {
  int i = blockIdx.x * blockDim.x + threadIdx.x;
  if (i >= 2112 * 64) return;
  int r = i >> 6, o = i & 63;
  float v;
  if (r < 2048) {
    int ii = r >> 5, k = r & 31;
    v = w2[l * 32 * 4096 + k * 4096 + ii * 64 + o];
  } else {
    v = b2[l * 4096 + (r - 2048) * 64 + o];
  }
  Bm[i] = (_Float16)v;
}

// ---- edge MLP first layer: z = relu(edge_attr @ W1 + b1), f16 [E,32] ----
__global__ void k_z(const float* __restrict__ ea, const float* __restrict__ w1,
                    const float* __restrict__ b1, _Float16* __restrict__ zh, int l, int E) {
  int i = blockIdx.x * blockDim.x + threadIdx.x;
  if (i >= E * 32) return;
  int e = i >> 5, j = i & 31;
  float acc = b1[l * 32 + j];
#pragma unroll
  for (int c = 0; c < 4; ++c) acc += ea[e * 4 + c] * w1[l * 128 + c * 32 + j];
  zh[i] = (_Float16)fmaxf(acc, 0.0f);
}

// ---- TDM: stage one contiguous 24KB group of Bm into LDS (1-D tile) ----
__device__ __forceinline__ void tdm_load_group(const _Float16* gsrc, unsigned lds_off) {
  unsigned long long ga = (unsigned long long)(uintptr_t)gsrc;
  // D# group 0: count=1 | lds_addr | global_addr[56:0] | type=2
  u32x4 g0 = { 1u,
               lds_off,
               (unsigned)(ga & 0xffffffffu),
               (unsigned)((ga >> 32) & 0x1ffffffu) | (2u << 30) };
  // D# group 1: data_size=8B, tensor_dim0=tile_dim0=GROUP_UNITS, dims1+=1/0
  i32x8 g1 = { (int)0x00030000,                                  // data_size=3 (8B)
               (int)((GROUP_UNITS & 0xffff) << 16),              // tensor_dim0[15:0]
               (int)(((GROUP_UNITS >> 16) & 0xffff) | (1 << 16)),// tensor_dim0[31:16] | tensor_dim1=1
               (int)((GROUP_UNITS & 0xffff) << 16),              // tile_dim0
               (int)1,                                           // tile_dim1=1, tile_dim2=0
               (int)GROUP_UNITS,                                 // tensor_dim0_stride[31:0]
               0, 0 };
  i32x4 z4 = { 0, 0, 0, 0 };
#if defined(__clang_major__) && (__clang_major__ >= 23)
  i32x8 z8 = { 0, 0, 0, 0, 0, 0, 0, 0 };
  __builtin_amdgcn_tensor_load_to_lds(g0, g1, z4, z4, z8, 0);
#else
  __builtin_amdgcn_tensor_load_to_lds(g0, g1, z4, z4, 0);
#endif
}

// ---- fused message GEMM + scatter, B staged in LDS via TDM double-buffer ----
__global__ void __launch_bounds__(256) k_msg(
    const _Float16* __restrict__ zh, const _Float16* __restrict__ hh,
    const _Float16* __restrict__ Bm, const int* __restrict__ src,
    const int* __restrict__ dst, float* __restrict__ agg, int ntiles) {
  __shared__ _Float16 Bl[2][GROUP_HALVES];   // 48 KB double buffer

  int wv = blockIdx.x * 8 + (threadIdx.x >> 5);
  int lane = threadIdx.x & 31;
  bool active = wv < ntiles;
  int tile = active ? wv : (ntiles - 1);     // clamp: keep all waves in the barriers
  int m = lane & 15, hi = lane >> 4;
  int e = tile * 16 + m;
  int s = src[e];

  // z fragment in 16x32 A layout
  V16 zf;
  const _Float16* zp = zh + (size_t)e * 32 + hi * 8;
  zf.h[0] = *(const v8h*)(zp);
  zf.h[1] = *(const v8h*)(zp + 16);
  const _Float16* hrow = hh + (size_t)s * 64;

  v8f c0 = {}, c1 = {}, c2 = {}, c3 = {};

  bool wave0 = (threadIdx.x >> 5) == 0;
  unsigned lds0 = (unsigned)(uintptr_t)(&Bl[0][0]);

  if (wave0) tdm_load_group(Bm, lds0);

  for (int g = 0; g < NGRP; ++g) {
    if (wave0) {
      if (g + 1 < NGRP) {
        tdm_load_group(Bm + (size_t)(g + 1) * GROUP_HALVES,
                       lds0 + (unsigned)(((g + 1) & 1) * GROUP_BYTES));
        __builtin_amdgcn_s_wait_tensorcnt(1);   // group g complete (in-order)
      } else {
        __builtin_amdgcn_s_wait_tensorcnt(0);
      }
    }
    __syncthreads();                            // publish buffer g&1 to all waves

    const _Float16* lbase = &Bl[g & 1][0];
    int kc0 = g * GRP;
#pragma unroll
    for (int cg = 0; cg < GRP; ++cg) {
      int kc = kc0 + cg;
      V16 a;
      if (kc < 64) {
        _Float16 hs = hrow[kc];
        a.v = zf.v * hs;                        // outer-product row: h_src[kc] * z
      } else {
        int c = kc - 64;                        // bias chunks: A = h_src directly
        const _Float16* hp = hrow + 32 * c + 8 * hi;
        a.h[0] = *(const v8h*)(hp);
        a.h[1] = *(const v8h*)(hp + 16);
      }
      V16 b0, b1, b2, b3;
#pragma unroll
      for (int t = 0; t < 16; ++t) {
        int kk = (t < 8) ? (8 * hi + t) : (16 + 8 * hi + (t - 8));
        const _Float16* bp = lbase + (size_t)(cg * 32 + kk) * 64 + m;
        b0.e[t] = bp[0]; b1.e[t] = bp[16]; b2.e[t] = bp[32]; b3.e[t] = bp[48];
      }
      c0 = __builtin_amdgcn_wmma_f32_16x16x32_f16(false, a.v, false, b0.v, (short)0, c0, false, false);
      c1 = __builtin_amdgcn_wmma_f32_16x16x32_f16(false, a.v, false, b1.v, (short)0, c1, false, false);
      c2 = __builtin_amdgcn_wmma_f32_16x16x32_f16(false, a.v, false, b2.v, (short)0, c2, false, false);
      c3 = __builtin_amdgcn_wmma_f32_16x16x32_f16(false, a.v, false, b3.v, (short)0, c3, false, false);
    }
    __syncthreads();                            // all reads done before buffer reuse
  }

  if (active) {
    // scatter-add: C layout -> row = r + 8*hi, col = m
    const int* dp = dst + tile * 16 + hi * 8;
    int dr[8];
#pragma unroll
    for (int r = 0; r < 8; ++r) dr[r] = dp[r];
#pragma unroll
    for (int r = 0; r < 8; ++r) {
      size_t base = (size_t)dr[r] * 64 + m;
      atomicAdd(&agg[base +  0], c0[r]);
      atomicAdd(&agg[base + 16], c1[r]);
      atomicAdd(&agg[base + 32], c2[r]);
      atomicAdd(&agg[base + 48], c3[r]);
    }
  }
}

// ---- combine: h2 = agg * inv_deg + h @ root_w + root_b ----
__global__ void k_comb(const float* __restrict__ agg, const float* __restrict__ invdeg,
                       const float* __restrict__ h, const float* __restrict__ rw,
                       const float* __restrict__ rb, float* __restrict__ h2, int l, int N) {
  int i = blockIdx.x * blockDim.x + threadIdx.x;
  if (i >= N * HD) return;
  int n = i >> 6, f = i & 63;
  float acc = agg[i] * invdeg[n] + rb[l * 64 + f];
#pragma unroll 8
  for (int j = 0; j < 64; ++j) acc += h[n * 64 + j] * rw[l * 4096 + j * 64 + f];
  h2[i] = acc;
}

// ---- per-feature sum & sumsq over nodes ----
__global__ void k_stats(const float* __restrict__ h2, float* __restrict__ st, int N) {
  __shared__ float s1[256], s2[256];
  int f = threadIdx.x & 63, g = threadIdx.x >> 6;
  float a1 = 0.f, a2 = 0.f;
  for (int n = blockIdx.x * 4 + g; n < N; n += gridDim.x * 4) {
    float v = h2[n * 64 + f];
    a1 += v; a2 += v * v;
  }
  s1[threadIdx.x] = a1; s2[threadIdx.x] = a2;
  __syncthreads();
  if (g == 0) {
    a1 = s1[f] + s1[64 + f] + s1[128 + f] + s1[192 + f];
    a2 = s2[f] + s2[64 + f] + s2[128 + f] + s2[192 + f];
    atomicAdd(&st[f], a1);
    atomicAdd(&st[64 + f], a2);
  }
}

// ---- GraphNorm (+LeakyReLU between layers); last layer writes d_out ----
__global__ void k_norm(const float* __restrict__ h2, const float* __restrict__ st,
                       const float* __restrict__ gw, const float* __restrict__ gb,
                       const float* __restrict__ gms, float* __restrict__ hout,
                       _Float16* __restrict__ hh, float* __restrict__ out,
                       int l, int N, int last) {
  int i = blockIdx.x * blockDim.x + threadIdx.x;
  if (i >= N * HD) return;
  int f = i & 63;
  float invN = 1.0f / (float)N;
  float mean = st[f] * invN;
  float ms = gms[l * 64 + f];
  // E[(h - ms*mean)^2] = E[h^2] - 2*ms*mean^2 + ms^2*mean^2
  float var = st[64 + f] * invN - 2.f * ms * mean * mean + ms * ms * mean * mean;
  float hc = h2[i] - ms * mean;
  float v = gw[l * 64 + f] * hc * rsqrtf(var + EPSV) + gb[l * 64 + f];
  if (last) {
    out[i] = v;
  } else {
    v = v > 0.f ? v : 0.2f * v;  // LeakyReLU(0.2)
    hout[i] = v;
    hh[i] = (_Float16)v;
  }
}

extern "C" void kernel_launch(void* const* d_in, const int* in_sizes, int n_in,
                              void* d_out, int out_size, void* d_ws, size_t ws_size,
                              hipStream_t stream) {
  const float* x   = (const float*)d_in[0];
  const int*   ei  = (const int*)d_in[1];
  const float* ea  = (const float*)d_in[2];
  const float* pw  = (const float*)d_in[3];
  const float* pb  = (const float*)d_in[4];
  const float* w1  = (const float*)d_in[5];
  const float* b1  = (const float*)d_in[6];
  const float* w2  = (const float*)d_in[7];
  const float* b2  = (const float*)d_in[8];
  const float* rw  = (const float*)d_in[9];
  const float* rb  = (const float*)d_in[10];
  const float* gw  = (const float*)d_in[11];
  const float* gb  = (const float*)d_in[12];
  const float* gms = (const float*)d_in[13];
  float* out = (float*)d_out;

  int N = in_sizes[0] / 8;
  int E = in_sizes[2] / 4;
  const int* src  = ei;
  const int* dstp = ei + E;

  char* ws = (char*)d_ws;
  size_t off = 0;
  auto alloc = [&](size_t bytes) -> void* {
    void* p = ws + off;
    off += (bytes + 255) & ~(size_t)255;
    return p;
  };
  float*    h      = (float*)alloc((size_t)N * 64 * 4);
  float*    h2     = (float*)alloc((size_t)N * 64 * 4);
  float*    agg    = (float*)alloc((size_t)N * 64 * 4);
  _Float16* hh     = (_Float16*)alloc((size_t)N * 64 * 2);
  _Float16* zh     = (_Float16*)alloc((size_t)E * 32 * 2);
  _Float16* Bm     = (_Float16*)alloc((size_t)2112 * 64 * 2);
  float*    invdeg = (float*)alloc((size_t)N * 4);
  float*    st     = (float*)alloc(512);

  hipMemsetAsync(invdeg, 0, (size_t)N * 4, stream);
  k_deg<<<(E + 255) / 256, 256, 0, stream>>>(dstp, invdeg, E);
  k_invdeg<<<(N + 255) / 256, 256, 0, stream>>>(invdeg, N);
  k_proj<<<(N * 64 + 255) / 256, 256, 0, stream>>>(x, pw, pb, h, hh, N);

  int ntiles = E / 16;  // E = 50000 -> 3125 exact
  for (int l = 0; l < 2; ++l) {
    k_prep<<<(2112 * 64 + 255) / 256, 256, 0, stream>>>(w2, b2, Bm, l);
    k_z<<<(E * 32 + 255) / 256, 256, 0, stream>>>(ea, w1, b1, zh, l, E);
    hipMemsetAsync(agg, 0, (size_t)N * 64 * 4, stream);
    k_msg<<<(ntiles + 7) / 8, 256, 0, stream>>>(zh, hh, Bm, src, dstp, agg, ntiles);
    k_comb<<<(N * 64 + 255) / 256, 256, 0, stream>>>(agg, invdeg, h, rw, rb, h2, l, N);
    hipMemsetAsync(st, 0, 512, stream);
    k_stats<<<64, 256, 0, stream>>>(h2, st, N);
    k_norm<<<(N * 64 + 255) / 256, 256, 0, stream>>>(h2, st, gw, gb, gms, h, hh, out,
                                                     l, N, l == 1 ? 1 : 0);
  }
}
